// HGCN_73495480369554
// MI455X (gfx1250) — compile-verified, matrix operand order checked by reading, not run on hardware
//
#include <hip/hip_runtime.h>

typedef __attribute__((ext_vector_type(16))) __bf16 bf16x16;
typedef __attribute__((ext_vector_type(8)))  __bf16 bf16x8;
typedef __attribute__((ext_vector_type(4)))  __bf16 bf16x4;
typedef __attribute__((ext_vector_type(8)))  float  v8f;
typedef unsigned int u32;
typedef __attribute__((ext_vector_type(4))) u32 u32x4;
typedef __attribute__((ext_vector_type(8))) int i32x8;
typedef __attribute__((ext_vector_type(4))) int i32x4;

#define PROJ_MAXN (1.0f - 4e-3f)
#define MIN_NORM  1e-15f
#define LDS_K     136   // 128 + 8 bf16 pad -> 272B row stride (16B aligned, conflict-free)

#if defined(__has_builtin)
#if __has_builtin(__builtin_amdgcn_tensor_load_to_lds)
#define HAVE_TDM 1
#endif
#endif

__device__ __forceinline__ float wave_sum(float v) {
#pragma unroll
  for (int off = 16; off > 0; off >>= 1) v += __shfl_xor(v, off, 32);
  return v;
}

__device__ __forceinline__ float artanh_c(float x) {
  x = fminf(fmaxf(x, -1.0f + 1e-7f), 1.0f - 1e-7f);
  return 0.5f * logf((1.0f + x) / (1.0f - x));
}

#ifdef HAVE_TDM
// ---------------------------------------------------------------------------
// TDM: 2D tile load, rows of 64 DWORDs (256B), LDS rows padded +16B (272B).
// D# per CDNA5 ISA ch.8: group0 {count, lds_addr, global_addr, type=2},
// group1 {data_size=4B, pad_interval=64dw(code 5), pad_amount=4dw(code 3),
//         tensor_dim0=64, tensor_dim1=rows_in_tensor, tile_dim0=64,
//         tile_dim1=tile_rows, tensor_dim0_stride=64}.
// OOB rows (beyond tensor_dim1) are zero-filled by the TDM.
// 6-arg builtin variant (clang-23 / therock-10.0 headers).
// ---------------------------------------------------------------------------
__device__ __forceinline__ void tdm_load_2d(u32 lds_off, const void* gptr,
                                            u32 tensor_rows, u32 tile_rows) {
  unsigned long long ga = (unsigned long long)gptr;
  u32x4 g0;
  g0[0] = 1u;                                            // count=1, user mode
  g0[1] = lds_off;                                       // LDS byte address
  g0[2] = (u32)(ga & 0xffffffffu);                       // global_addr[31:0]
  g0[3] = (u32)((ga >> 32) & 0x01ffffffu) | (2u << 30);  // addr[56:32] | type=2

  const u32 dim0 = 64u, tile0 = 64u, stride0 = 64u;      // DWORD units
  i32x8 g1;
  g1[0] = (int)((2u << 16)      // data_size = 4B
              | (1u << 20)      // pad_enable
              | (5u << 22)      // pad_interval: 64 DWORDs
              | (3u << 25));    // pad_amount: 4 DWORDs (16B)
  g1[1] = (int)(dim0 << 16);                                   // tensor_dim0 lo
  g1[2] = (int)((dim0 >> 16) | ((tensor_rows & 0xffffu) << 16));
  g1[3] = (int)((tensor_rows >> 16) | (tile0 << 16));          // tile_dim0
  g1[4] = (int)(tile_rows & 0xffffu);                          // tile_dim1 (dim2=0)
  g1[5] = (int)stride0;                                        // dim0_stride lo
  g1[6] = 0;
  g1[7] = 0;
  i32x4 z4 = (i32x4){0, 0, 0, 0};
  i32x8 z8 = (i32x8){0, 0, 0, 0, 0, 0, 0, 0};
  __builtin_amdgcn_tensor_load_to_lds(g0, g1, z4, z4, z8, 0);
}
#endif

// ---------------------------------------------------------------------------
// K1: convert W1, W2 (f32) -> bf16
// ---------------------------------------------------------------------------
__global__ __launch_bounds__(256) void convert_w_kernel(
    const float* __restrict__ W1, const float* __restrict__ W2,
    __bf16* __restrict__ Wbf) {
  int i = blockIdx.x * 256 + threadIdx.x;
  if (i < 16384)       Wbf[i] = (__bf16)W1[i];
  else if (i < 32768)  Wbf[i] = (__bf16)W2[i - 16384];
}

// ---------------------------------------------------------------------------
// K0: encode  xh = proj(expmap0(x,1),1); emit bf16 rows + row norms
// ---------------------------------------------------------------------------
__global__ __launch_bounds__(256) void encode_kernel(
    const float* __restrict__ x, __bf16* __restrict__ Xbf,
    float* __restrict__ xn, int N) {
  int row  = blockIdx.x * 8 + (threadIdx.x >> 5);
  if (row >= N) return;
  int lane = threadIdx.x & 31;

  float4 u = ((const float4*)(x + (size_t)row * 128))[lane];
  float ss = u.x * u.x + u.y * u.y + u.z * u.z + u.w * u.w;
  float nrm = fmaxf(sqrtf(wave_sum(ss)), MIN_NORM);
  float t   = tanhf(nrm);
  float s   = t / nrm;
  float ps  = (t > PROJ_MAXN) ? (PROJ_MAXN / t) : 1.0f;
  s *= ps;

  bf16x4 o;
  o[0] = (__bf16)(u.x * s); o[1] = (__bf16)(u.y * s);
  o[2] = (__bf16)(u.z * s); o[3] = (__bf16)(u.w * s);
  ((bf16x4*)(Xbf + (size_t)row * 128))[lane] = o;
  if (lane == 0) xn[row] = fmaxf(t * ps, MIN_NORM);
}

// ---------------------------------------------------------------------------
// K2: GEMM  mx = Xbf @ Wbf^T ; TDM-staged tiles, v_wmma_f32_16x16x32_bf16
// ---------------------------------------------------------------------------
__global__ __launch_bounds__(256) void gemm_wmma_kernel(
    const __bf16* __restrict__ X, const __bf16* __restrict__ W,
    float* __restrict__ out, int N) {
  __shared__ __bf16 Xs[128 * LDS_K];
  __shared__ __bf16 Ws[128 * LDS_K];

  const int tid     = threadIdx.x;
  const int lane    = tid & 31;
  const int wave    = tid >> 5;
  const int rowBase = blockIdx.x * 128;

#ifdef HAVE_TDM
  if (tid == 0) {
    u32 xs_off = (u32)((unsigned long long)(void*)Xs & 0xffffffffull);
    u32 ws_off = (u32)((unsigned long long)(void*)Ws & 0xffffffffull);
    tdm_load_2d(xs_off, X + (size_t)rowBase * 128, (u32)(N - rowBase), 128u);
    tdm_load_2d(ws_off, W, 128u, 128u);
  }
  __builtin_amdgcn_s_wait_tensorcnt(0);
  __syncthreads();
#else
  {
    const int r = tid >> 1;
    const int h = (tid & 1) * 64;
    const uint4* gw = (const uint4*)(W + r * 128 + h);
    uint4*       lw = (uint4*)(Ws + r * LDS_K + h);
#pragma unroll
    for (int i = 0; i < 8; ++i) lw[i] = gw[i];

    uint4* lx = (uint4*)(Xs + r * LDS_K + h);
    int gr = rowBase + r;
    if (gr < N) {
      const uint4* gx = (const uint4*)(X + (size_t)gr * 128 + h);
#pragma unroll
      for (int i = 0; i < 8; ++i) lx[i] = gx[i];
    } else {
      uint4 z = make_uint4(0u, 0u, 0u, 0u);
#pragma unroll
      for (int i = 0; i < 8; ++i) lx[i] = z;
    }
  }
  __syncthreads();
#endif

  const int half    = lane >> 4;   // K-half selector
  const int rsub    = lane & 15;   // row (A) / col (B) within tile
  const int waveRow = wave * 16;

  v8f acc[8];
#pragma unroll
  for (int n = 0; n < 8; ++n) acc[n] = (v8f){0.f,0.f,0.f,0.f,0.f,0.f,0.f,0.f};

#pragma unroll
  for (int k = 0; k < 128; k += 32) {
    bf16x16 a;
    {
      const __bf16* ap = Xs + (waveRow + rsub) * LDS_K + k + half * 8;
      bf16x8 lo = *(const bf16x8*)(ap);
      bf16x8 hi = *(const bf16x8*)(ap + 16);
#pragma unroll
      for (int i = 0; i < 8; ++i) { a[i] = lo[i]; a[i + 8] = hi[i]; }
    }
#pragma unroll
    for (int n = 0; n < 8; ++n) {
      bf16x16 b;
      const __bf16* bp = Ws + (n * 16 + rsub) * LDS_K + k + half * 8;
      bf16x8 lo = *(const bf16x8*)(bp);
      bf16x8 hi = *(const bf16x8*)(bp + 16);
#pragma unroll
      for (int i = 0; i < 8; ++i) { b[i] = lo[i]; b[i + 8] = hi[i]; }
      acc[n] = __builtin_amdgcn_wmma_f32_16x16x32_bf16(
          false, a, false, b, (short)0, acc[n], false, false);
    }
  }

  // C layout: VGPR i -> M = waveRow + i + 8*half ; N = n*16 + rsub
  const int outRow0 = rowBase + waveRow + half * 8;
#pragma unroll
  for (int n = 0; n < 8; ++n) {
#pragma unroll
    for (int i = 0; i < 8; ++i) {
      int r = outRow0 + i;
      if (r < N) out[(size_t)r * 128 + n * 16 + rsub] = acc[n][i];
    }
  }
}

// ---------------------------------------------------------------------------
// K3: post-linear: mobius_matvec tail + proj + hyperbolic bias + proj +
//     logmap0 -> XT; re-zeroes M row (it becomes the agg buffer).
// ---------------------------------------------------------------------------
__global__ __launch_bounds__(256) void post_linear_kernel(
    float* __restrict__ M, const float* __restrict__ xn,
    const float* __restrict__ bvec, float* __restrict__ XT, int N) {
  int row  = blockIdx.x * 8 + (threadIdx.x >> 5);
  if (row >= N) return;
  int lane = threadIdx.x & 31;
  float* mrow = M + (size_t)row * 128;

  float4 mx = ((const float4*)mrow)[lane];
  float mxn = fmaxf(sqrtf(wave_sum(mx.x*mx.x + mx.y*mx.y + mx.z*mx.z + mx.w*mx.w)), MIN_NORM);
  float x_n = xn[row];

  float t = tanhf(mxn / x_n * artanh_c(x_n));
  float s = t / mxn;
  float rn = fabsf(t);
  float ps = (rn > PROJ_MAXN) ? (PROJ_MAXN / rn) : 1.0f;
  s *= ps;
  float4 h;
  h.x = mx.x * s; h.y = mx.y * s; h.z = mx.z * s; h.w = mx.w * s;

  float4 bb = ((const float4*)bvec)[lane];
  float bn = fmaxf(sqrtf(wave_sum(bb.x*bb.x + bb.y*bb.y + bb.z*bb.z + bb.w*bb.w)), MIN_NORM);
  float bt = tanhf(bn);
  float bs = bt / bn;
  float bps = (bt > PROJ_MAXN) ? (PROJ_MAXN / bt) : 1.0f;
  bs *= bps;
  float4 hb;
  hb.x = bb.x * bs; hb.y = bb.y * bs; hb.z = bb.z * bs; hb.w = bb.w * bs;

  float x2 = wave_sum(h.x*h.x + h.y*h.y + h.z*h.z + h.w*h.w);
  float y2 = wave_sum(hb.x*hb.x + hb.y*hb.y + hb.z*hb.z + hb.w*hb.w);
  float xy = wave_sum(h.x*hb.x + h.y*hb.y + h.z*hb.z + h.w*hb.w);
  float na = 1.0f + 2.0f * xy + y2;
  float nb = 1.0f - x2;
  float den = fmaxf(1.0f + 2.0f * xy + x2 * y2, MIN_NORM);
  float inv = 1.0f / den;
  float4 m;
  m.x = (na*h.x + nb*hb.x) * inv; m.y = (na*h.y + nb*hb.y) * inv;
  m.z = (na*h.z + nb*hb.z) * inv; m.w = (na*h.w + nb*hb.w) * inv;

  float mn = fmaxf(sqrtf(wave_sum(m.x*m.x + m.y*m.y + m.z*m.z + m.w*m.w)), MIN_NORM);
  float mps = (mn > PROJ_MAXN) ? (PROJ_MAXN / mn) : 1.0f;
  float mn2 = fmaxf(mn * mps, MIN_NORM);
  float lt = artanh_c(mn2) / mn2 * mps;
  float4 xt;
  xt.x = m.x * lt; xt.y = m.y * lt; xt.z = m.z * lt; xt.w = m.w * lt;
  ((float4*)(XT + (size_t)row * 128))[lane] = xt;

  float4 z = make_float4(0.f, 0.f, 0.f, 0.f);
  ((float4*)mrow)[lane] = z;
}

// ---------------------------------------------------------------------------
// K4: edge scatter  agg[dst] += ew * XT[src]; one wave per edge, fp32 atomics
// ---------------------------------------------------------------------------
__global__ __launch_bounds__(256) void scatter_kernel(
    const float* __restrict__ XT, const int* __restrict__ src,
    const int* __restrict__ dst, const float* __restrict__ ew,
    float* __restrict__ M, int E) {
  int e = blockIdx.x * 8 + (threadIdx.x >> 5);
  if (e >= E) return;
  int lane = threadIdx.x & 31;
  int sN = src[e], dN = dst[e];
  float w = ew[e];
  float4 v = ((const float4*)(XT + (size_t)sN * 128))[lane];
  float* p = M + (size_t)dN * 128 + lane * 4;
  unsafeAtomicAdd(p + 0, w * v.x);
  unsafeAtomicAdd(p + 1, w * v.y);
  unsafeAtomicAdd(p + 2, w * v.z);
  unsafeAtomicAdd(p + 3, w * v.w);
}

// ---------------------------------------------------------------------------
// K5: post-agg: proj(expmap0(agg)) -> relu(logmap0) -> proj(expmap0)
// ---------------------------------------------------------------------------
__global__ __launch_bounds__(256) void post_agg_kernel(
    const float* __restrict__ M, __bf16* __restrict__ Xbf,
    float* __restrict__ xn, float* __restrict__ outf, int N, int final_layer) {
  int row  = blockIdx.x * 8 + (threadIdx.x >> 5);
  if (row >= N) return;
  int lane = threadIdx.x & 31;

  float4 a = ((const float4*)(M + (size_t)row * 128))[lane];
  float an = fmaxf(sqrtf(wave_sum(a.x*a.x + a.y*a.y + a.z*a.z + a.w*a.w)), MIN_NORM);
  float t  = tanhf(an);
  float s  = t / an;
  float ps = (t > PROJ_MAXN) ? (PROJ_MAXN / t) : 1.0f;
  s *= ps;
  float hn = fmaxf(t * ps, MIN_NORM);

  float lt = artanh_c(hn) / hn;
  float4 xt;
  xt.x = fmaxf(a.x * s * lt, 0.f); xt.y = fmaxf(a.y * s * lt, 0.f);
  xt.z = fmaxf(a.z * s * lt, 0.f); xt.w = fmaxf(a.w * s * lt, 0.f);

  float on = fmaxf(sqrtf(wave_sum(xt.x*xt.x + xt.y*xt.y + xt.z*xt.z + xt.w*xt.w)), MIN_NORM);
  float ot = tanhf(on);
  float os = ot / on;
  float ops = (ot > PROJ_MAXN) ? (PROJ_MAXN / ot) : 1.0f;
  os *= ops;
  float4 o;
  o.x = xt.x * os; o.y = xt.y * os; o.z = xt.z * os; o.w = xt.w * os;

  if (final_layer) {
    ((float4*)(outf + (size_t)row * 128))[lane] = o;
  } else {
    bf16x4 ob;
    ob[0] = (__bf16)o.x; ob[1] = (__bf16)o.y; ob[2] = (__bf16)o.z; ob[3] = (__bf16)o.w;
    ((bf16x4*)(Xbf + (size_t)row * 128))[lane] = ob;
    if (lane == 0) xn[row] = fmaxf(ot * ops, MIN_NORM);
  }
}

// ---------------------------------------------------------------------------
extern "C" void kernel_launch(void* const* d_in, const int* in_sizes, int n_in,
                              void* d_out, int out_size, void* d_ws, size_t ws_size,
                              hipStream_t stream) {
  const float* x  = (const float*)d_in[0];
  const float* W1 = (const float*)d_in[1];
  const float* b1 = (const float*)d_in[2];
  const float* W2 = (const float*)d_in[3];
  const float* b2 = (const float*)d_in[4];
  const float* ew = (const float*)d_in[5];
  const int*  src = (const int*)d_in[6];
  const int*  dst = (const int*)d_in[7];
  float* out = (float*)d_out;

  const int N = in_sizes[0] / 128;   // 100000
  const int E = in_sizes[5];         // 1600000

  char* ws = (char*)d_ws;
  size_t off = 0;
  auto carve = [&](size_t bytes) -> void* {
    void* p = ws + off;
    off += (bytes + 255) & ~(size_t)255;
    return p;
  };
  __bf16* Xbf = (__bf16*)carve((size_t)N * 128 * 2);
  __bf16* Wbf = (__bf16*)carve((size_t)2 * 128 * 128 * 2);
  float*  xn  = (float*)carve((size_t)N * 4);
  float*  M   = (float*)carve((size_t)N * 128 * 4);   // mx / agg (shared)
  float*  XT  = (float*)carve((size_t)N * 128 * 4);

  const int rowBlocks  = (N + 7) / 8;
  const int gemmBlocks = (N + 127) / 128;
  const int edgeBlocks = (E + 7) / 8;

  convert_w_kernel<<<128, 256, 0, stream>>>(W1, W2, Wbf);
  encode_kernel<<<rowBlocks, 256, 0, stream>>>(x, Xbf, xn, N);

  // ----- layer 1 -----
  gemm_wmma_kernel<<<gemmBlocks, 256, 0, stream>>>(Xbf, Wbf, M, N);
  post_linear_kernel<<<rowBlocks, 256, 0, stream>>>(M, xn, b1, XT, N);
  scatter_kernel<<<edgeBlocks, 256, 0, stream>>>(XT, src, dst, ew, M, E);
  post_agg_kernel<<<rowBlocks, 256, 0, stream>>>(M, Xbf, xn, out, N, 0);

  // ----- layer 2 -----
  gemm_wmma_kernel<<<gemmBlocks, 256, 0, stream>>>(Xbf, Wbf + 16384, M, N);
  post_linear_kernel<<<rowBlocks, 256, 0, stream>>>(M, xn, b2, XT, N);
  scatter_kernel<<<edgeBlocks, 256, 0, stream>>>(XT, src, dst, ew, M, E);
  post_agg_kernel<<<rowBlocks, 256, 0, stream>>>(M, Xbf, xn, out, N, 1);
}
